// AdvancedTransformerModel_49108656062990
// MI455X (gfx1250) — matrix operand order repaired and checked by reading, not compiled, not run on hardware
//
#include <hip/hip_runtime.h>
#include <math.h>

typedef _Float16 h16;
typedef _Float16 v8h  __attribute__((ext_vector_type(8)));
typedef _Float16 v16h __attribute__((ext_vector_type(16)));
typedef float    v8f  __attribute__((ext_vector_type(8)));
typedef unsigned int v4u __attribute__((ext_vector_type(4)));
typedef int          v4i __attribute__((ext_vector_type(4)));
typedef int          v8i __attribute__((ext_vector_type(8)));

constexpr int Bc = 2, Sc = 1024, Dc = 768, Hc = 12, DHc = 64;
constexpr int Lc = 2, Ec = 8, Fc = 3072, Vc = 32000;
constexpr int BSc = Bc * Sc;     // 2048 tokens
constexpr int CAP = BSc;         // per-expert token slot capacity

#if defined(__gfx1250__) && __has_builtin(__builtin_amdgcn_tensor_load_to_lds)
#define USE_TDM 1
#else
#define USE_TDM 0
#endif

static __device__ __forceinline__ v8f wmma16(v16h a, v16h b, v8f c) {
  return __builtin_amdgcn_wmma_f32_16x16x32_f16(false, a, false, b, (short)0, c, false, false);
}
static __device__ __forceinline__ v16h cat8(v8h lo, v8h hi) {
  return __builtin_shufflevector(lo, hi, 0,1,2,3,4,5,6,7,8,9,10,11,12,13,14,15);
}
static __device__ __forceinline__ v8f zero8() {
  v8f z = {0.f,0.f,0.f,0.f,0.f,0.f,0.f,0.f};
  return z;
}

// ---------------------------------------------------------------------------
// Generic f16 WMMA GEMM: C[M,N] = A[M,K] * BT[N,K]^T (+bias) with epilogues.
// Block: 256 threads = 8 waves (4x2), block tile 128x128, K-step 32.
// B tile staged to LDS by the Tensor Data Mover (TDM); A tile (token-gathered
// for MoE) staged on the vector path.
// ---------------------------------------------------------------------------
enum { EPI_STORE = 0, EPI_GELU = 1, EPI_RESADD = 2, EPI_SCATTER = 3 };

template <int EPI, bool GATHER>
__global__ __launch_bounds__(256) void gemm_f16_wmma(
    const h16* __restrict__ A, const h16* __restrict__ BT,
    const float* __restrict__ bias,
    float* __restrict__ Cf, h16* __restrict__ Ch,
    const int* __restrict__ rows_list, const float* __restrict__ rows_wgt,
    const int* __restrict__ count_ptr,
    int M, int N, int K, int ldc) {
  int cnt = count_ptr ? *count_ptr : M;
  const int m0 = blockIdx.y * 128;
  if (m0 >= cnt) return;                       // block-uniform early exit
  const int n0 = blockIdx.x * 128;

  __shared__ __align__(16) h16 sA[128][40];    // 32 K-halfs + 8 pad
  __shared__ __align__(16) h16 sB[128][40];    // TDM pads rows to 40 halfs

  const int tid  = threadIdx.x;
  const int lane = tid & 31;
  const int wave = tid >> 5;
  const int wm   = wave >> 1;                  // 0..3 (M direction)
  const int wn   = wave & 1;                   // 0..1 (N direction)
  const int rl   = lane & 15;
  const int aOff = (lane < 16) ? 0 : 8;        // A frag: K interleave select
  const int bOff = (lane < 16) ? 0 : 8;        // B frag: K interleave select
  const int mhalf = (lane < 16) ? 0 : 8;

  v8f acc[2][4];
#pragma unroll
  for (int i = 0; i < 2; ++i)
#pragma unroll
    for (int j = 0; j < 4; ++j) acc[i][j] = zero8();

  const int ldsRow = tid >> 2;                 // 0..63
  const int ldsCol = (tid & 3) * 8;            // 0,8,16,24

#if USE_TDM
  // Invariant parts of the Tensor DMA descriptor (D#) for the B tile:
  // 128 rows x 64B (32 halfs), 8-byte elements, row stride K halfs in memory,
  // LDS pad of 4 DWORDs after every 16 DWORDs -> 40-half LDS row stride.
  const unsigned ldsB = (unsigned)(size_t)&sB[0][0];
  const unsigned d0u  = (unsigned)(K >> 2);    // row length in 8B units
  v8i g1;
  g1[0] = (int)((3u << 16) |                   // data_size = 8B
                (1u << 20) |                   // pad_enable
                (3u << 22) |                   // pad_interval: 16 DWORDs
                (3u << 25));                   // pad_amount: 4 DWORDs
  g1[1] = (int)((d0u & 0xFFFFu) << 16);        // tensor_dim0[15:0]
  g1[2] = (int)(((d0u >> 16) & 0xFFFFu) | (128u << 16)); // dim0 hi | tensor_dim1 lo
  g1[3] = (int)(8u << 16);                     // tensor_dim1 hi=0 | tile_dim0=8
  g1[4] = (int)128u;                           // tile_dim1=128, tile_dim2=0
  g1[5] = (int)d0u;                            // tensor_dim0_stride lo (8B units)
  g1[6] = 0;                                   // stride0 hi | stride1 lo
  g1[7] = 0;                                   // stride1 hi
  v4i z4 = {0, 0, 0, 0};
#endif

  for (int k0 = 0; k0 < K; k0 += 32) {
#if USE_TDM
    if (wave == 0) {
      unsigned long long gaddr =
          (unsigned long long)(size_t)(BT + (size_t)n0 * K + k0);
      v4u g0;
      g0.x = 1u;                               // count=1 user descriptor
      g0.y = ldsB;                             // LDS byte address
      g0.z = (unsigned)gaddr;                  // global_addr[31:0]
      g0.w = (unsigned)((gaddr >> 32) & 0x1FFFFFFu) | (2u << 30); // addr hi|type=2
#if __clang_major__ >= 23
      v8i z8 = {0, 0, 0, 0, 0, 0, 0, 0};
      __builtin_amdgcn_tensor_load_to_lds(g0, g1, z4, z4, z8, 0);
#else
      __builtin_amdgcn_tensor_load_to_lds(g0, g1, z4, z4, 0);
#endif
    }
#endif
#pragma unroll
    for (int it = 0; it < 2; ++it) {
      int row = ldsRow + it * 64;
      int gm = m0 + row;
      int src_m;
      if (GATHER) src_m = (gm < cnt) ? rows_list[gm] : 0;
      else        src_m = (gm < M) ? gm : 0;
      *(v8h*)&sA[row][ldsCol] = *(const v8h*)(A + (size_t)src_m * K + k0 + ldsCol);
#if !USE_TDM
      *(v8h*)&sB[row][ldsCol] = *(const v8h*)(BT + (size_t)(n0 + row) * K + k0 + ldsCol);
#endif
    }
#if USE_TDM
    if (wave == 0) __builtin_amdgcn_s_wait_tensorcnt(0);
#endif
    __syncthreads();

    v16h af[2], bf[4];
#pragma unroll
    for (int i = 0; i < 2; ++i) {
      const h16* p = &sA[wm * 32 + i * 16 + rl][0];
      af[i] = cat8(*(const v8h*)(p + aOff), *(const v8h*)(p + aOff + 16));
    }
#pragma unroll
    for (int j = 0; j < 4; ++j) {
      const h16* p = &sB[wn * 64 + j * 16 + rl][0];
      bf[j] = cat8(*(const v8h*)(p + bOff), *(const v8h*)(p + bOff + 16));
    }
#pragma unroll
    for (int i = 0; i < 2; ++i)
#pragma unroll
      for (int j = 0; j < 4; ++j)
        acc[i][j] = wmma16(af[i], bf[j], acc[i][j]);
    __syncthreads();
  }

#pragma unroll
  for (int i = 0; i < 2; ++i) {
#pragma unroll
    for (int r = 0; r < 8; ++r) {
      int m = m0 + wm * 32 + i * 16 + r + mhalf;
      bool valid = (m < cnt);
      int tok = 0; float wgt = 0.f;
      if (EPI == EPI_SCATTER && valid) { tok = rows_list[m]; wgt = rows_wgt[m]; }
#pragma unroll
      for (int j = 0; j < 4; ++j) {
        int n = n0 + wn * 64 + j * 16 + rl;
        float v = acc[i][j][r];
        if (bias) v += bias[n];
        if (EPI == EPI_STORE) {
          if (valid) Cf[(size_t)m * ldc + n] = v;
        } else if (EPI == EPI_RESADD) {
          if (valid) Cf[(size_t)m * ldc + n] = Cf[(size_t)m * ldc + n] + v;
        } else if (EPI == EPI_GELU) {
          if (valid) Ch[(size_t)m * ldc + n] =
              (h16)(0.5f * v * (1.f + erff(v * 0.70710678118654752f)));
        } else {  // EPI_SCATTER: weighted combine into residual stream
          if (valid) atomicAdd(&Cf[(size_t)tok * ldc + n], wgt * v);
        }
      }
    }
  }
}

// ---------------------------------------------------------------------------
// Flash attention: one wave per 16 query rows per (b,h). Causal, online softmax.
// Q,K: [B*H, S, 64] f16 ; VT: [B*H, 64, S] f16 ; ctx out: [B*S, D] f16
// ---------------------------------------------------------------------------
__global__ __launch_bounds__(128) void flash_attn(
    const h16* __restrict__ Q, const h16* __restrict__ Kh,
    const h16* __restrict__ VT, h16* __restrict__ ctx) {
  __shared__ __align__(16) h16 pbuf[4][16][72];   // per-wave P tile
  const int lane = threadIdx.x & 31;
  const int wv   = threadIdx.x >> 5;
  const int wid  = blockIdx.x * 4 + wv;
  const int MT   = Sc / 16;
  const int mt   = wid % MT;
  const int bh   = wid / MT;
  const int m0   = mt * 16;
  const int rl   = lane & 15;
  const int aOff = (lane < 16) ? 0 : 8;
  const int bOff = (lane < 16) ? 0 : 8;
  const int mhalf = (lane < 16) ? 0 : 8;

  const h16* qb = Q  + (size_t)bh * Sc * DHc;
  const h16* kb = Kh + (size_t)bh * Sc * DHc;
  const h16* vb = VT + (size_t)bh * DHc * Sc;

  v16h qf[2];
#pragma unroll
  for (int kk = 0; kk < 2; ++kk) {
    const h16* p = qb + (size_t)(m0 + rl) * DHc + kk * 32;
    qf[kk] = cat8(*(const v8h*)(p + aOff), *(const v8h*)(p + aOff + 16));
  }

  v8f o[4];
#pragma unroll
  for (int j = 0; j < 4; ++j) o[j] = zero8();
  float mrow[8], lrow[8];
#pragma unroll
  for (int r = 0; r < 8; ++r) { mrow[r] = -1e30f; lrow[r] = 0.f; }

  for (int j0 = 0; j0 < m0 + 16; j0 += 64) {
    v8f sc[4];
#pragma unroll
    for (int nt = 0; nt < 4; ++nt) sc[nt] = zero8();
#pragma unroll
    for (int nt = 0; nt < 4; ++nt) {
      int col = j0 + nt * 16 + rl;
#pragma unroll
      for (int kk = 0; kk < 2; ++kk) {
        const h16* p = kb + (size_t)col * DHc + kk * 32;
        v16h bf = cat8(*(const v8h*)(p + bOff), *(const v8h*)(p + bOff + 16));
        sc[nt] = wmma16(qf[kk], bf, sc[nt]);
      }
    }
#pragma unroll
    for (int r = 0; r < 8; ++r) {
      int m = m0 + r + mhalf;
      float tmax = -1e30f;
#pragma unroll
      for (int nt = 0; nt < 4; ++nt) {
        int j = j0 + nt * 16 + rl;
        float s = sc[nt][r] * 0.125f;              // 1/sqrt(64)
        if (j > m) s = -1e30f;                     // causal mask
        sc[nt][r] = s;
        tmax = fmaxf(tmax, s);
      }
#pragma unroll
      for (int off = 1; off <= 8; off <<= 1)
        tmax = fmaxf(tmax, __shfl_xor(tmax, off, 32));
      float newm = fmaxf(mrow[r], tmax);
      float corr = __expf(mrow[r] - newm);
      float psum = 0.f;
#pragma unroll
      for (int nt = 0; nt < 4; ++nt) {
        float pv = __expf(sc[nt][r] - newm);
        psum += pv;
        pbuf[wv][r + mhalf][nt * 16 + rl] = (h16)pv;
      }
#pragma unroll
      for (int off = 1; off <= 8; off <<= 1) psum += __shfl_xor(psum, off, 32);
      lrow[r] = lrow[r] * corr + psum;
      mrow[r] = newm;
#pragma unroll
      for (int nt = 0; nt < 4; ++nt) o[nt][r] *= corr;
    }
    asm volatile("s_wait_dscnt 0" ::: "memory");   // LDS write->read, same wave
    v16h pf[2];
#pragma unroll
    for (int kk = 0; kk < 2; ++kk) {
      const h16* p = &pbuf[wv][rl][kk * 32];
      pf[kk] = cat8(*(const v8h*)(p + aOff), *(const v8h*)(p + aOff + 16));
    }
#pragma unroll
    for (int nt = 0; nt < 4; ++nt) {
      int d = nt * 16 + rl;
#pragma unroll
      for (int kk = 0; kk < 2; ++kk) {
        const h16* p = vb + (size_t)d * Sc + j0 + kk * 32;
        v16h bf = cat8(*(const v8h*)(p + bOff), *(const v8h*)(p + bOff + 16));
        o[nt] = wmma16(pf[kk], bf, o[nt]);
      }
    }
  }
  const int b = bh / Hc, h = bh % Hc;
#pragma unroll
  for (int nt = 0; nt < 4; ++nt) {
#pragma unroll
    for (int r = 0; r < 8; ++r) {
      int m = m0 + r + mhalf;
      int d = nt * 16 + rl;
      ctx[((size_t)(b * Sc + m)) * Dc + h * DHc + d] = (h16)(o[nt][r] / lrow[r]);
    }
  }
}

// ---------------------------------------------------------------------------
// LayerNorm: one wave per row; writes f32 (optional) + f16 outputs.
// ---------------------------------------------------------------------------
__global__ __launch_bounds__(256) void layernorm_k(
    const float* __restrict__ x, const float* __restrict__ g,
    const float* __restrict__ b, float* __restrict__ yf, h16* __restrict__ yh) {
  const int lane = threadIdx.x & 31;
  const int wv = threadIdx.x >> 5;
  const int row = blockIdx.x * 8 + wv;
  if (row >= BSc) return;
  const float* xr = x + (size_t)row * Dc;
  float v[Dc / 32];
  float s = 0.f, s2 = 0.f;
#pragma unroll
  for (int i = 0; i < Dc / 32; ++i) {
    v[i] = xr[lane + i * 32];
    s += v[i]; s2 += v[i] * v[i];
  }
#pragma unroll
  for (int off = 1; off <= 16; off <<= 1) {
    s += __shfl_xor(s, off, 32);
    s2 += __shfl_xor(s2, off, 32);
  }
  float mean = s * (1.f / Dc);
  float var = s2 * (1.f / Dc) - mean * mean;
  float inv = rsqrtf(var + 1e-5f);
#pragma unroll
  for (int i = 0; i < Dc / 32; ++i) {
    int c = lane + i * 32;
    float y = (v[i] - mean) * inv * g[c] + b[c];
    if (yf) yf[(size_t)row * Dc + c] = y;
    yh[(size_t)row * Dc + c] = (h16)y;
  }
}

// ---------------------------------------------------------------------------
// Embedding: x = tok_emb[ids] + pos_emb
// ---------------------------------------------------------------------------
__global__ void embed_k(const int* __restrict__ ids, const float* __restrict__ tok,
                        const float* __restrict__ pos, float* __restrict__ x) {
  int i = blockIdx.x * 256 + threadIdx.x;
  if (i >= BSc * Dc) return;
  int t = i / Dc, c = i % Dc;
  int s = t % Sc;
  x[i] = tok[(size_t)ids[t] * Dc + c] + pos[(size_t)s * Dc + c];
}

// ---------------------------------------------------------------------------
// Split fused QKV (f32) into per-head f16 Q, K and transposed V.
// ---------------------------------------------------------------------------
__global__ void split_qkv_k(const float* __restrict__ qkv, h16* __restrict__ q,
                            h16* __restrict__ k, h16* __restrict__ vT) {
  int i = blockIdx.x * 256 + threadIdx.x;
  if (i >= BSc * Dc) return;
  int t = i / Dc, hd = i % Dc;
  int h = hd / DHc, d = hd % DHc;
  int b = t / Sc, s = t % Sc;
  size_t bh = (size_t)(b * Hc + h);
  const float* src = qkv + (size_t)t * (3 * Dc);
  q[(bh * Sc + s) * DHc + d]  = (h16)src[hd];
  k[(bh * Sc + s) * DHc + d]  = (h16)src[Dc + hd];
  vT[(bh * DHc + d) * Sc + s] = (h16)src[2 * Dc + hd];
}

// ---------------------------------------------------------------------------
// Router: one wave per token; softmax over 8 experts, top-2, renorm, append to
// per-expert token lists (fixed-capacity, counts in device memory).
// ---------------------------------------------------------------------------
__global__ __launch_bounds__(256) void router_k(
    const float* __restrict__ xn, const float* __restrict__ Rw,
    int* __restrict__ counts, int* __restrict__ tlist, float* __restrict__ twgt) {
  const int lane = threadIdx.x & 31;
  const int wv = threadIdx.x >> 5;
  const int t = blockIdx.x * 8 + wv;
  if (t >= BSc) return;
  float acc[Ec];
#pragma unroll
  for (int e = 0; e < Ec; ++e) acc[e] = 0.f;
  const float* xr = xn + (size_t)t * Dc;
  for (int d = lane; d < Dc; d += 32) {
    float xv = xr[d];
    const float* rw = Rw + (size_t)d * Ec;
#pragma unroll
    for (int e = 0; e < Ec; ++e) acc[e] += xv * rw[e];
  }
#pragma unroll
  for (int e = 0; e < Ec; ++e)
#pragma unroll
    for (int off = 1; off <= 16; off <<= 1) acc[e] += __shfl_xor(acc[e], off, 32);
  if (lane == 0) {
    float mx = acc[0];
    for (int e = 1; e < Ec; ++e) mx = fmaxf(mx, acc[e]);
    float p[Ec]; float sum = 0.f;
    for (int e = 0; e < Ec; ++e) { p[e] = __expf(acc[e] - mx); sum += p[e]; }
    for (int e = 0; e < Ec; ++e) p[e] /= sum;
    int e0 = 0;
    for (int e = 1; e < Ec; ++e) if (p[e] > p[e0]) e0 = e;
    int e1 = (e0 == 0) ? 1 : 0;
    for (int e = 0; e < Ec; ++e) if (e != e0 && p[e] > p[e1]) e1 = e;
    float den = p[e0] + p[e1];
    int s0 = atomicAdd(&counts[e0], 1);
    tlist[e0 * CAP + s0] = t; twgt[e0 * CAP + s0] = p[e0] / den;
    int s1 = atomicAdd(&counts[e1], 1);
    tlist[e1 * CAP + s1] = t; twgt[e1 * CAP + s1] = p[e1] / den;
  }
}

// ---------------------------------------------------------------------------
// Transpose + convert: dst[C,R] (f16) = src[R,C] (f32)
// ---------------------------------------------------------------------------
__global__ __launch_bounds__(256) void transpose_cvt(
    const float* __restrict__ src, h16* __restrict__ dst, int R, int C) {
  __shared__ float tile[32][33];
  int r0 = blockIdx.y * 32, c0 = blockIdx.x * 32;
  int tx = threadIdx.x % 32, ty = threadIdx.x / 32;  // 32 x 8
#pragma unroll
  for (int i = 0; i < 32; i += 8) {
    int r = r0 + ty + i, c = c0 + tx;
    tile[ty + i][tx] = (r < R && c < C) ? src[(size_t)r * C + c] : 0.f;
  }
  __syncthreads();
#pragma unroll
  for (int i = 0; i < 32; i += 8) {
    int c = c0 + ty + i, r = r0 + tx;
    if (c < C && r < R) dst[(size_t)c * R + r] = (h16)tile[tx][ty + i];
  }
}

// ---------------------------------------------------------------------------
extern "C" void kernel_launch(void* const* d_in, const int* in_sizes, int n_in,
                              void* d_out, int out_size, void* d_ws, size_t ws_size,
                              hipStream_t stream) {
  (void)in_sizes; (void)n_in; (void)out_size; (void)ws_size;
  const int*   input_ids = (const int*)  d_in[0];
  const float* tok_emb   = (const float*)d_in[1];
  const float* pos_emb   = (const float*)d_in[2];
  const float* ln1_g     = (const float*)d_in[3];
  const float* ln1_b     = (const float*)d_in[4];
  const float* wqkv      = (const float*)d_in[5];
  const float* bqkv      = (const float*)d_in[6];
  const float* wo        = (const float*)d_in[7];
  const float* bo        = (const float*)d_in[8];
  const float* ln2_g     = (const float*)d_in[9];
  const float* ln2_b     = (const float*)d_in[10];
  const float* router_w  = (const float*)d_in[11];
  const float* w1        = (const float*)d_in[12];
  const float* b1        = (const float*)d_in[13];
  const float* w2        = (const float*)d_in[14];
  const float* b2        = (const float*)d_in[15];
  const float* lnf_g     = (const float*)d_in[16];
  const float* lnf_b     = (const float*)d_in[17];
  const float* wout      = (const float*)d_in[18];

  char* base = (char*)d_ws;
  size_t off = 0;
  auto alloc = [&](size_t bytes) -> void* {
    void* r = base + off;
    off = (off + bytes + 255) & ~(size_t)255;
    return r;
  };
  float* x      = (float*)alloc((size_t)BSc * Dc * 4);
  float* xn_f   = (float*)alloc((size_t)BSc * Dc * 4);
  h16*   xn_h   = (h16*)  alloc((size_t)BSc * Dc * 2);
  float* qkv    = (float*)alloc((size_t)BSc * 3 * Dc * 4);
  h16*   qh     = (h16*)  alloc((size_t)BSc * Dc * 2);
  h16*   kh     = (h16*)  alloc((size_t)BSc * Dc * 2);
  h16*   vTh    = (h16*)  alloc((size_t)BSc * Dc * 2);
  h16*   ctxh   = (h16*)  alloc((size_t)BSc * Dc * 2);
  h16*   wqkvT  = (h16*)  alloc((size_t)3 * Dc * Dc * 2);
  h16*   woT    = (h16*)  alloc((size_t)Dc * Dc * 2);
  h16*   w1T    = (h16*)  alloc((size_t)Ec * Fc * Dc * 2);
  h16*   w2T    = (h16*)  alloc((size_t)Ec * Dc * Fc * 2);
  h16*   woutT  = (h16*)  alloc((size_t)Vc * Dc * 2);
  h16*   hbuf   = (h16*)  alloc((size_t)Ec * CAP * Fc * 2);
  int*   counts = (int*)  alloc(Ec * sizeof(int));
  int*   tlist  = (int*)  alloc((size_t)Ec * CAP * sizeof(int));
  float* twgt   = (float*)alloc((size_t)Ec * CAP * sizeof(float));

  dim3 b256(256);
  transpose_cvt<<<dim3(Vc / 32, Dc / 32), b256, 0, stream>>>(wout, woutT, Dc, Vc);
  embed_k<<<(BSc * Dc + 255) / 256, b256, 0, stream>>>(input_ids, tok_emb, pos_emb, x);

  for (int l = 0; l < Lc; ++l) {
    transpose_cvt<<<dim3(3 * Dc / 32, Dc / 32), b256, 0, stream>>>(
        wqkv + (size_t)l * Dc * 3 * Dc, wqkvT, Dc, 3 * Dc);
    transpose_cvt<<<dim3(Dc / 32, Dc / 32), b256, 0, stream>>>(
        wo + (size_t)l * Dc * Dc, woT, Dc, Dc);
    for (int e = 0; e < Ec; ++e) {
      transpose_cvt<<<dim3(Fc / 32, Dc / 32), b256, 0, stream>>>(
          w1 + ((size_t)l * Ec + e) * Dc * Fc, w1T + (size_t)e * Fc * Dc, Dc, Fc);
      transpose_cvt<<<dim3(Dc / 32, Fc / 32), b256, 0, stream>>>(
          w2 + ((size_t)l * Ec + e) * Fc * Dc, w2T + (size_t)e * Dc * Fc, Fc, Dc);
    }
    // LN1 -> f16 activations
    layernorm_k<<<BSc / 8, b256, 0, stream>>>(x, ln1_g + l * Dc, ln1_b + l * Dc,
                                              nullptr, xn_h);
    // QKV projection
    gemm_f16_wmma<EPI_STORE, false><<<dim3(3 * Dc / 128, BSc / 128), b256, 0, stream>>>(
        xn_h, wqkvT, bqkv + (size_t)l * 3 * Dc, qkv, nullptr, nullptr, nullptr, nullptr,
        BSc, 3 * Dc, Dc, 3 * Dc);
    split_qkv_k<<<(BSc * Dc + 255) / 256, b256, 0, stream>>>(qkv, qh, kh, vTh);
    flash_attn<<<Bc * Hc * (Sc / 16) / 4, dim3(128), 0, stream>>>(qh, kh, vTh, ctxh);
    // x += ctx @ Wo + bo
    gemm_f16_wmma<EPI_RESADD, false><<<dim3(Dc / 128, BSc / 128), b256, 0, stream>>>(
        ctxh, woT, bo + (size_t)l * Dc, x, nullptr, nullptr, nullptr, nullptr,
        BSc, Dc, Dc, Dc);
    // LN2 -> f32 (router) + f16 (experts)
    layernorm_k<<<BSc / 8, b256, 0, stream>>>(x, ln2_g + l * Dc, ln2_b + l * Dc,
                                              xn_f, xn_h);
    hipMemsetAsync(counts, 0, Ec * sizeof(int), stream);
    router_k<<<BSc / 8, b256, 0, stream>>>(xn_f, router_w + (size_t)l * Dc * Ec,
                                           counts, tlist, twgt);
    for (int e = 0; e < Ec; ++e) {
      // FC1 (gathered rows) + exact GELU -> f16 hidden
      gemm_f16_wmma<EPI_GELU, true><<<dim3(Fc / 128, CAP / 128), b256, 0, stream>>>(
          xn_h, w1T + (size_t)e * Fc * Dc, b1 + ((size_t)l * Ec + e) * Fc,
          nullptr, hbuf + (size_t)e * CAP * Fc,
          tlist + e * CAP, nullptr, counts + e,
          CAP, Fc, Dc, Fc);
      // FC2 + weighted scatter-add into residual stream
      gemm_f16_wmma<EPI_SCATTER, false><<<dim3(Dc / 128, CAP / 128), b256, 0, stream>>>(
          hbuf + (size_t)e * CAP * Fc, w2T + (size_t)e * Dc * Fc,
          b2 + ((size_t)l * Ec + e) * Dc,
          x, nullptr,
          tlist + e * CAP, twgt + e * CAP, counts + e,
          CAP, Dc, Fc, Dc);
    }
  }
  // Final LN + logits
  layernorm_k<<<BSc / 8, b256, 0, stream>>>(x, lnf_g, lnf_b, nullptr, xn_h);
  gemm_f16_wmma<EPI_STORE, false><<<dim3(Vc / 128, BSc / 128), b256, 0, stream>>>(
      xn_h, woutT, nullptr, (float*)d_out, nullptr, nullptr, nullptr, nullptr,
      BSc, Vc, Dc, Vc);
}
